// RNNAttention_72524817760555
// MI455X (gfx1250) — matrix operand order repaired and checked by reading, not compiled, not run on hardware
//
#include <hip/hip_runtime.h>
#include <math.h>

// Problem constants (from reference): N=128, S=256, T1=65 (T=64 steps),
// H=D=512, 4H=2048, gates-K = D+H+H = 1536.
#define NN   128
#define SS   256
#define T1C  65
#define TT   64
#define HH   512
#define DD   512
#define G4H  2048
#define KG   1536
#define NEG_INF (-1000000000.0f)

typedef __attribute__((ext_vector_type(16))) __bf16 v16bf;
typedef __attribute__((ext_vector_type(8)))  __bf16 v8bf;
typedef __attribute__((ext_vector_type(4)))  __bf16 v4bf;
typedef __attribute__((ext_vector_type(8)))  float  v8f;

// ---------------------------------------------------------------------------
// fp32 -> bf16 bulk convert (grid-stride)
// ---------------------------------------------------------------------------
__global__ void cvt_bf16_kernel(const float* __restrict__ in,
                                __bf16* __restrict__ out, int total) {
    for (int i = blockIdx.x * blockDim.x + threadIdx.x; i < total;
         i += gridDim.x * blockDim.x)
        out[i] = (__bf16)in[i];
}

// ---------------------------------------------------------------------------
// Pack a row-major fp32 (K x Ncols) weight matrix into the per-lane WMMA
// bf16 B-fragment layout: frag id = (k0/32)*NT + nt ; within a frag,
// lane l holds col n = nt*16 + (l&15), K values k = k0 + (l>>4)*16 + j,
// j = 0..15 contiguous per lane (32B per lane).
// ---------------------------------------------------------------------------
__global__ void pack_b_kernel(const float* __restrict__ W,
                              __bf16* __restrict__ out, int K, int Ncols) {
    int gid = blockIdx.x * blockDim.x + threadIdx.x;
    if (gid >= K * Ncols) return;
    int NT = Ncols >> 4;
    int fid = gid >> 9, r = gid & 511, lane = r >> 4, j = r & 15;
    int kt = fid / NT, nt = fid % NT;
    int k = (kt << 5) + ((lane >> 4) << 4) + j;
    int n = (nt << 4) + (lane & 15);
    out[gid] = (__bf16)W[(size_t)k * Ncols + n];
}

// Stacked gates weight [Wx ; Wh ; W_context] -> packed bf16, K=1536, N=2048.
__global__ void pack_b_stack_kernel(const float* __restrict__ Wx,
                                    const float* __restrict__ Wh,
                                    const float* __restrict__ Wc,
                                    __bf16* __restrict__ out) {
    int gid = blockIdx.x * blockDim.x + threadIdx.x;
    if (gid >= KG * G4H) return;
    const int NT = G4H >> 4;
    int fid = gid >> 9, r = gid & 511, lane = r >> 4, j = r & 15;
    int kt = fid / NT, nt = fid % NT;
    int k = (kt << 5) + ((lane >> 4) << 4) + j;
    int n = (nt << 4) + (lane & 15);
    const float* src; int kk;
    if (k < DD)            { src = Wx; kk = k; }
    else if (k < DD + HH)  { src = Wh; kk = k - DD; }
    else                   { src = Wc; kk = k - DD - HH; }
    out[gid] = (__bf16)src[(size_t)kk * G4H + n];
}

// ---------------------------------------------------------------------------
// Embedding gather: Aall[t][n][0:512] = bf16(W_embed[captions[n][t]])
// Aall rows are KG=1536 wide: [ x_t | h | ctx ].
// ---------------------------------------------------------------------------
__global__ void embed_kernel(const int* __restrict__ captions,
                             const float* __restrict__ Wemb,
                             __bf16* __restrict__ aall) {
    int t = blockIdx.x, n = blockIdx.y;
    int row = captions[n * T1C + t];
    const float* src = Wemb + (size_t)row * DD;
    __bf16* dst = aall + ((size_t)t * NN + n) * KG;
    for (int d = threadIdx.x; d < DD; d += blockDim.x)
        dst[d] = (__bf16)src[d];
}

// ---------------------------------------------------------------------------
// bf16 WMMA GEMM, K/NCOLS compile-time so all fragment addresses fold to
// immediate offsets. One wave computes a 16x32 tile (two adjacent N-tiles
// sharing the A fragment): two independent v_wmma per fragment-load group,
// half the A traffic, better latency overlap. B-stream prefetched via
// __builtin_prefetch (-> global_prefetch_b8).
// ---------------------------------------------------------------------------
template <int K, int NCOLS>
__global__ void gemm_bf16_wmma(const __bf16* __restrict__ A, int lda,
                               const __bf16* __restrict__ Bp,
                               float* __restrict__ outF,
                               __bf16* __restrict__ outB,
                               const float* __restrict__ bias,
                               int M) {
    constexpr int NT = NCOLS / 16;   // N-tiles
    constexpr int NP = NT / 2;       // N-tile pairs per M-tile
    constexpr int KC = K / 32;       // K chunks
    int wave = blockIdx.x * (blockDim.x >> 5) + (threadIdx.x >> 5);
    if (wave >= (M >> 4) * NP) return;      // wave-uniform: EXEC all-ones
    int mt = wave / NP, nt0 = (wave % NP) * 2;
    int lane = threadIdx.x & 31;
    int kHi  = lane >> 4;                    // 0 or 1 (lane half)
    const __bf16* arow = A + (size_t)((mt << 4) + (lane & 15)) * lda;
    // per-lane base into packed B for (kc=0, nt0); per-kc stride = NT*512,
    // adjacent N-tile = +512 halves.
    const __bf16* brow = Bp + ((size_t)nt0 * 32 + lane) * 16;

    v8f acc0 = {}, acc1 = {};
#pragma unroll 4
    for (int kc = 0; kc < KC; ++kc) {
        // A fragment: 16 contiguous bf16 per lane, k = kHi*16 + j (matches packed B)
        v8bf a0 = *(const v8bf*)(arow + kc * 32 + kHi * 16);
        v8bf a1 = *(const v8bf*)(arow + kc * 32 + kHi * 16 + 8);
        v16bf af;
#pragma unroll
        for (int i = 0; i < 8; ++i) { af[i] = a0[i]; af[i + 8] = a1[i]; }
        // B fragments: 32B contiguous per lane, constant offsets
        const __bf16* bp = brow + (size_t)kc * NT * 512;
        v16bf b0 = *(const v16bf*)bp;
        v16bf b1 = *(const v16bf*)(bp + 512);
        if (kc + 2 < KC) {
            __builtin_prefetch(bp + (size_t)2 * NT * 512, 0, 1);
            __builtin_prefetch(bp + (size_t)2 * NT * 512 + 512, 0, 1);
        }
        acc0 = __builtin_amdgcn_wmma_f32_16x16x32_bf16(
            false, af, false, b0, (short)0, acc0, false, false);
        acc1 = __builtin_amdgcn_wmma_f32_16x16x32_bf16(
            false, af, false, b1, (short)0, acc1, false, false);
    }

    int n0 = (nt0 << 4) + (lane & 15);
    float bv0 = bias ? bias[n0] : 0.0f;
    float bv1 = bias ? bias[n0 + 16] : 0.0f;
#pragma unroll
    for (int r = 0; r < 8; ++r) {
        int m = (mt << 4) + r + (kHi << 3);
        float v0 = acc0[r] + bv0;
        float v1 = acc1[r] + bv1;
        if (outB) {
            outB[(size_t)m * NCOLS + n0]      = (__bf16)v0;
            outB[(size_t)m * NCOLS + n0 + 16] = (__bf16)v1;
        } else {
            outF[(size_t)m * NCOLS + n0]      = v0;
            outF[(size_t)m * NCOLS + n0 + 16] = v1;
        }
    }
}

// ---------------------------------------------------------------------------
// Attention scores: e[n,s] = sum_h tanh(dec[n,h] + encpb[n,s,h]) * wa2[h]
// (b_attention1 pre-added into encpb). One wave per (n,s). Mask applied.
// ---------------------------------------------------------------------------
__global__ void attn_scores_kernel(const float* __restrict__ dec,
                                   const __bf16* __restrict__ encpb,
                                   const float* __restrict__ wa2,
                                   const float* __restrict__ mask,
                                   float* __restrict__ e) {
    int n = blockIdx.y;
    int s = (blockIdx.x << 3) + (threadIdx.x >> 5);
    int lane = threadIdx.x & 31;
    const float* dp = dec + (size_t)n * HH;
    const __bf16* ep = encpb + ((size_t)n * SS + s) * HH;
    float sum = 0.0f;
#pragma unroll
    for (int i = 0; i < 4; ++i) {
        int h = i * 128 + lane * 4;
        v4bf ev = *(const v4bf*)(ep + h);
        float4 dv = *(const float4*)(dp + h);
        float4 wv = *(const float4*)(wa2 + h);
        sum += tanhf(dv.x + (float)ev[0]) * wv.x;
        sum += tanhf(dv.y + (float)ev[1]) * wv.y;
        sum += tanhf(dv.z + (float)ev[2]) * wv.z;
        sum += tanhf(dv.w + (float)ev[3]) * wv.w;
    }
#pragma unroll
    for (int off = 16; off > 0; off >>= 1) sum += __shfl_xor(sum, off, 32);
    if (lane == 0)
        e[n * SS + s] = (mask[n * SS + s] > 0.0f) ? sum : NEG_INF;
}

// ---------------------------------------------------------------------------
// Row softmax over S=256 (one block of 256 threads per row).
// ---------------------------------------------------------------------------
__global__ void softmax256_kernel(const float* __restrict__ e,
                                  float* __restrict__ alpha) {
    __shared__ float red[256];
    int n = blockIdx.x, s = threadIdx.x;
    float v = e[n * SS + s];
    red[s] = v; __syncthreads();
    for (int off = 128; off > 0; off >>= 1) {
        if (s < off) red[s] = fmaxf(red[s], red[s + off]);
        __syncthreads();
    }
    float mx = red[0]; __syncthreads();
    float ex = __expf(v - mx);
    red[s] = ex; __syncthreads();
    for (int off = 128; off > 0; off >>= 1) {
        if (s < off) red[s] += red[s + off];
        __syncthreads();
    }
    alpha[n * SS + s] = ex / red[0];
}

// ---------------------------------------------------------------------------
// ctx[n,h] = sum_s alpha[n,s] * encb[n,s,h]  -> written as bf16 into the
// ctx-slice of this step's activation rows Aall[t][n][1024:1536].
// ---------------------------------------------------------------------------
__global__ void ctx_kernel(const float* __restrict__ alpha,
                           const __bf16* __restrict__ encb,
                           __bf16* __restrict__ aall_t) {
    int n = blockIdx.y;
    int h = (blockIdx.x << 8) + threadIdx.x;
    const __bf16* ep = encb + (size_t)n * SS * HH + h;
    const float* al = alpha + n * SS;
    float acc = 0.0f;
    for (int s = 0; s < SS; ++s) acc += al[s] * (float)ep[(size_t)s * HH];
    aall_t[(size_t)n * KG + DD + HH + h] = (__bf16)acc;
}

// ---------------------------------------------------------------------------
// LSTM pointwise: gates row layout [i | f | o | g] of 4H=2048.
// Writes: c (fp32 state), hs output (N,T,H fp32), hb (bf16 for dec_proj A),
// and next step's h-slice of Aall.
// ---------------------------------------------------------------------------
__global__ void lstm_kernel(const float* __restrict__ gates,
                            float* __restrict__ c,
                            float* __restrict__ hsOut,
                            __bf16* __restrict__ hb,
                            __bf16* __restrict__ aall_next, int t) {
    int n = blockIdx.y;
    int h = (blockIdx.x << 8) + threadIdx.x;
    const float* g = gates + (size_t)n * G4H;
    float iv = 1.0f / (1.0f + __expf(-g[h]));
    float fv = 1.0f / (1.0f + __expf(-g[HH + h]));
    float ov = 1.0f / (1.0f + __expf(-g[2 * HH + h]));
    float gv = tanhf(g[3 * HH + h]);
    float cn = fv * c[n * HH + h] + iv * gv;
    float hn = ov * tanhf(cn);
    c[n * HH + h] = cn;
    hsOut[((size_t)n * TT + t) * HH + h] = hn;
    hb[n * HH + h] = (__bf16)hn;
    if (aall_next) aall_next[(size_t)n * KG + DD + h] = (__bf16)hn;
}

// Initial state: c=0, hb = bf16(h_init), Aall[0] h-slice = bf16(h_init).
__global__ void init_kernel(const float* __restrict__ h_init,
                            float* __restrict__ c,
                            __bf16* __restrict__ hb,
                            __bf16* __restrict__ aall0) {
    int n = blockIdx.y;
    int h = (blockIdx.x << 8) + threadIdx.x;
    float v = h_init[n * HH + h];
    c[n * HH + h] = 0.0f;
    hb[n * HH + h] = (__bf16)v;
    aall0[(size_t)n * KG + DD + h] = (__bf16)v;
}

// ---------------------------------------------------------------------------
extern "C" void kernel_launch(void* const* d_in, const int* in_sizes, int n_in,
                              void* d_out, int out_size, void* d_ws, size_t ws_size,
                              hipStream_t stream) {
    const float* enc      = (const float*)d_in[0];   // (N,S,H)
    const int*   captions = (const int*)  d_in[1];   // (N,T1)
    const float* h_init   = (const float*)d_in[2];   // (N,H)
    const float* mask     = (const float*)d_in[3];   // (N,S)
    const float* Wemb     = (const float*)d_in[4];   // (V,D)
    const float* Wx       = (const float*)d_in[5];   // (D,4H)
    const float* Wh       = (const float*)d_in[6];   // (H,4H)
    const float* b        = (const float*)d_in[7];   // (4H)
    const float* Watt1    = (const float*)d_in[8];   // (2H,H)
    const float* b_att1   = (const float*)d_in[9];   // (H)
    const float* Wa2      = (const float*)d_in[10];  // (H,1)
    const float* Wctx     = (const float*)d_in[11];  // (H,4H)
    float* hsOut          = (float*)d_out;           // (N,T,H)

    // ---- workspace suballocation (all offsets 256B aligned) ---------------
    char* ws = (char*)d_ws;
    size_t off = 0;
    auto suballoc = [&](size_t bytes) -> void* {
        void* p = ws + off;
        off += (bytes + 255) & ~(size_t)255;
        return p;
    };
    __bf16* encb    = (__bf16*)suballoc((size_t)NN * SS * HH * 2);   // 33.6 MB (L2-resident)
    __bf16* encpb   = (__bf16*)suballoc((size_t)NN * SS * HH * 2);   // 33.6 MB (L2-resident)
    __bf16* Aall    = (__bf16*)suballoc((size_t)TT * NN * KG * 2);   // 25.2 MB
    __bf16* BpGates = (__bf16*)suballoc((size_t)KG * G4H * 2);       //  6.3 MB
    __bf16* BpEnc   = (__bf16*)suballoc((size_t)HH * HH * 2);
    __bf16* BpDec   = (__bf16*)suballoc((size_t)HH * HH * 2);
    __bf16* hb      = (__bf16*)suballoc((size_t)NN * HH * 2);
    float*  dec     = (float*) suballoc((size_t)NN * HH * 4);
    float*  e       = (float*) suballoc((size_t)NN * SS * 4);
    float*  alpha   = (float*) suballoc((size_t)NN * SS * 4);
    float*  gates   = (float*) suballoc((size_t)NN * G4H * 4);
    float*  cstate  = (float*) suballoc((size_t)NN * HH * 4);

    // ---- precompute (step-invariant) --------------------------------------
    cvt_bf16_kernel<<<16384, 256, 0, stream>>>(enc, encb, NN * SS * HH);
    pack_b_kernel<<<(HH * HH + 255) / 256, 256, 0, stream>>>(
        Watt1 + (size_t)HH * HH, BpEnc, HH, HH);          // W_attention1[H:]
    pack_b_kernel<<<(HH * HH + 255) / 256, 256, 0, stream>>>(
        Watt1, BpDec, HH, HH);                            // W_attention1[:H]
    pack_b_stack_kernel<<<(KG * G4H + 255) / 256, 256, 0, stream>>>(
        Wx, Wh, Wctx, BpGates);
    embed_kernel<<<dim3(TT, NN), 256, 0, stream>>>(captions, Wemb, Aall);
    // enc_proj (+ b_attention1) in bf16: (N*S x H) @ (H x H)
    // waves = 2048 M-tiles * 16 pairs = 32768 -> 8192 blocks of 4 waves
    gemm_bf16_wmma<HH, HH><<<8192, 128, 0, stream>>>(
        encb, HH, BpEnc, nullptr, encpb, b_att1, NN * SS);
    init_kernel<<<dim3(2, NN), 256, 0, stream>>>(h_init, cstate, hb, Aall);

    // ---- sequential recurrence --------------------------------------------
    for (int t = 0; t < TT; ++t) {
        __bf16* aall_t    = Aall + (size_t)t * NN * KG;
        __bf16* aall_next = (t + 1 < TT) ? Aall + (size_t)(t + 1) * NN * KG
                                         : nullptr;
        // dec_proj = h @ W1_dec : 128x512x512, 8*16=128 waves -> 32 blocks
        gemm_bf16_wmma<HH, HH><<<32, 128, 0, stream>>>(
            hb, HH, BpDec, dec, nullptr, nullptr, NN);
        // scores + mask
        attn_scores_kernel<<<dim3(SS / 8, NN), 256, 0, stream>>>(
            dec, encpb, Wa2, mask, e);
        // softmax over S
        softmax256_kernel<<<NN, 256, 0, stream>>>(e, alpha);
        // ctx -> Aall[t] ctx-slice (bf16)
        ctx_kernel<<<dim3(2, NN), 256, 0, stream>>>(alpha, encb, aall_t);
        // gates = [x|h|ctx] @ [Wx;Wh;Wc] + b : 128x1536x2048,
        // 8*64=512 waves -> 128 blocks
        gemm_bf16_wmma<KG, G4H><<<128, 128, 0, stream>>>(
            aall_t, KG, BpGates, gates, nullptr, b, NN);
        // LSTM pointwise update; writes hs[t], hb, next h-slice, c
        lstm_kernel<<<dim3(2, NN), 256, 0, stream>>>(
            gates, cstate, hsOut, hb, aall_next, t);
    }
}
